// AxialShift_6511170421750
// MI455X (gfx1250) — compile-verified, matrix operand order checked by reading, not run on hardware
//
#include <hip/hip_runtime.h>
#include <hip/hip_bf16.h>
#include <math.h>
#include <stdint.h>

// ---------------------------------------------------------------------------
// Shift-MLP block on MI455X (gfx1250, wave32): bf16 WMMA GEMMs fed by the
// Tensor Data Mover into double-buffered LDS.
// ---------------------------------------------------------------------------

typedef __bf16 bf16_t;
typedef __attribute__((ext_vector_type(16))) __bf16 v16bf;
typedef __attribute__((ext_vector_type(8)))  float  v8f;
typedef unsigned int u32x4 __attribute__((ext_vector_type(4)));
typedef int          i32x4 __attribute__((ext_vector_type(4)));
typedef int          i32x8 __attribute__((ext_vector_type(8)));

#define DIMC 768
#define NPIX 784          // 28*28
#define NB   32
#define HH   28
#define WW   28
#define SLAB (DIMC * NPIX)  // 602112 elements per sample

__device__ __forceinline__ float gelu_exact(float x) {
    return 0.5f * x * (1.0f + erff(x * 0.70710678118654752440f));
}

union FragBF {
    v16bf v;
    uint4 u4[2];
};

// ---------------------------------------------------------------------------
// TDM: 2D tile load Global -> LDS (bf16 elements, data_size code 1 = 2 bytes).
// D# field packing per CDNA5 ISA ch.8 (group0: count|lds_addr|global_addr|type,
// group1: data_size|tensor dims|tile dims|stride). Groups 2/3 zero (<=2D).
// ---------------------------------------------------------------------------
__device__ __forceinline__ void tdm_load_2d(uint64_t gaddr, uint32_t lds_off,
                                            uint32_t tile_d0, uint32_t tile_d1,
                                            uint32_t tensor_d0, uint32_t tensor_d1,
                                            uint32_t stride0)
{
    u32x4 g0;
    g0[0] = 1u;                                  // count=1, user descriptor
    g0[1] = lds_off;                             // LDS byte address
    g0[2] = (uint32_t)gaddr;                     // global_addr[31:0]
    g0[3] = ((uint32_t)(gaddr >> 32) & 0x01FFFFFFu) | (2u << 30); // addr[56:32] | type=2

    i32x8 g1;
    g1[0] = (int)(1u << 16);                     // wg_mask=0, data_size=1 (2B), no pad/iter
    g1[1] = (int)((tensor_d0 & 0xFFFFu) << 16);  // atomic_barrier=0 | tensor_dim0[15:0]
    g1[2] = (int)((tensor_d0 >> 16) | ((tensor_d1 & 0xFFFFu) << 16)); // td0 hi | td1 lo
    g1[3] = (int)((tensor_d1 >> 16) | (tile_d0 << 16));               // td1 hi | tile_dim0
    g1[4] = (int)(tile_d1 & 0xFFFFu);            // tile_dim1 | tile_dim2=0
    g1[5] = (int)stride0;                        // tensor_dim0_stride[31:0]
    g1[6] = 0;                                   // stride0[47:32] | stride1[15:0]
    g1[7] = 0;                                   // stride1[47:16]

    i32x4 z4 = {0, 0, 0, 0};
#if __clang_major__ >= 23
    i32x8 z8 = {0, 0, 0, 0, 0, 0, 0, 0};
    __builtin_amdgcn_tensor_load_to_lds(g0, g1, z4, z4, z8, 0);
#else
    __builtin_amdgcn_tensor_load_to_lds(g0, g1, z4, z4, 0);
#endif
}

// ---------------------------------------------------------------------------
// WMMA GEMM:  Out[b][m][n] = epilogue( sum_k Wt[m][k] * Xt[b][n][k] + bias[m] )
//   Wt : DIMC x DIMC bf16 row-major (M x K)
//   Xt : per batch NPIX x DIMC bf16 row-major (N x K)  [K-major activations]
//   Out: per batch DIMC x NPIX fp32 (spatial layout)
// EPI: 0 = bias, 1 = bias+gelu, 2 = bias+gelu then accumulate into Out
//
// Block = 7 waves. All waves share one 64-row weight slab; wave w owns n-tile
// (n_group*7 + w). Wave 0 drives a TDM double-buffer pipeline: per 32-wide
// K-step it DMAs a 64x32 weight tile and a 112x32 activation tile into LDS
// while all waves run 4 WMMAs per step out of the other buffer via ds_load.
// ---------------------------------------------------------------------------
template <int EPI>
__global__ __launch_bounds__(224) void gemm_wmma_k(
    const bf16_t* __restrict__ Wt,
    const bf16_t* __restrict__ Xt,
    const float*  __restrict__ bias,
    float* __restrict__ Out)
{
    __shared__ alignas(16) bf16_t shA[2][64 * 32];    // weight tiles   (4 KB each)
    __shared__ alignas(16) bf16_t shB[2][112 * 32];   // activation tiles (7 KB each)

    const int wave = threadIdx.x >> 5;     // 0..6
    const int lane = threadIdx.x & 31;
    const int half = lane >> 4;            // K-half of the fragment
    const int lr   = lane & 15;            // row (A: M) / col (B: N) within tile

    const int m_super = blockIdx.x % 12;   // 768 / 64
    const int n_group = blockIdx.x / 12;   // 0..6
    const int m0      = m_super * 64;
    const int n_base  = n_group * 112;     // 7 tiles of 16
    const int n0      = n_base + wave * 16;
    const int b       = blockIdx.y;

    const uint64_t wt_base = (uint64_t)(uintptr_t)(Wt + (size_t)m0 * DIMC);
    const uint64_t xt_base = (uint64_t)(uintptr_t)(Xt + (size_t)b * NPIX * DIMC +
                                                   (size_t)n_base * DIMC);

    v8f acc0 = {}, acc1 = {}, acc2 = {}, acc3 = {};

    // ---- prologue: fill buffer 0 with the k=0 tiles -------------------------
    if (threadIdx.x < 32) {
        tdm_load_2d(wt_base, (uint32_t)(uintptr_t)&shA[0][0], 32, 64, DIMC, DIMC, DIMC);
        tdm_load_2d(xt_base, (uint32_t)(uintptr_t)&shB[0][0], 32, 112, DIMC, NPIX, DIMC);
        __builtin_amdgcn_s_wait_tensorcnt(0);
    }
    __syncthreads();

    for (int k0 = 0; k0 < DIMC; k0 += 32) {
        const int cur = (k0 >> 5) & 1;

        // prefetch next K-step tiles into the other buffer (overlaps compute)
        if (threadIdx.x < 32 && (k0 + 32) < DIMC) {
            tdm_load_2d(wt_base + (size_t)(k0 + 32) * 2,
                        (uint32_t)(uintptr_t)&shA[cur ^ 1][0], 32, 64, DIMC, DIMC, DIMC);
            tdm_load_2d(xt_base + (size_t)(k0 + 32) * 2,
                        (uint32_t)(uintptr_t)&shB[cur ^ 1][0], 32, 112, DIMC, NPIX, DIMC);
        }

        const int kb0 = half * 8;          // lane's first 8 bf16 within the 32-K tile
        const int kb1 = kb0 + 16;

        FragBF bfrag;
        {
            const bf16_t* brow = &shB[cur][(wave * 16 + lr) * 32];
            bfrag.u4[0] = *(const uint4*)(brow + kb0);
            bfrag.u4[1] = *(const uint4*)(brow + kb1);
        }

        FragBF a0, a1, a2, a3;
        {
            const bf16_t* arow = &shA[cur][lr * 32];
            a0.u4[0] = *(const uint4*)(arow + kb0);
            a0.u4[1] = *(const uint4*)(arow + kb1);
            a1.u4[0] = *(const uint4*)(arow + 16 * 32 + kb0);
            a1.u4[1] = *(const uint4*)(arow + 16 * 32 + kb1);
            a2.u4[0] = *(const uint4*)(arow + 32 * 32 + kb0);
            a2.u4[1] = *(const uint4*)(arow + 32 * 32 + kb1);
            a3.u4[0] = *(const uint4*)(arow + 48 * 32 + kb0);
            a3.u4[1] = *(const uint4*)(arow + 48 * 32 + kb1);
        }

        acc0 = __builtin_amdgcn_wmma_f32_16x16x32_bf16(false, a0.v, false, bfrag.v,
                                                       (short)0, acc0, false, false);
        acc1 = __builtin_amdgcn_wmma_f32_16x16x32_bf16(false, a1.v, false, bfrag.v,
                                                       (short)0, acc1, false, false);
        acc2 = __builtin_amdgcn_wmma_f32_16x16x32_bf16(false, a2.v, false, bfrag.v,
                                                       (short)0, acc2, false, false);
        acc3 = __builtin_amdgcn_wmma_f32_16x16x32_bf16(false, a3.v, false, bfrag.v,
                                                       (short)0, acc3, false, false);

        // make the prefetched buffer visible before anyone flips to it
        if (threadIdx.x < 32) {
            __builtin_amdgcn_s_wait_tensorcnt(0);
        }
        __syncthreads();
    }

    v8f accs[4] = {acc0, acc1, acc2, acc3};
    float* obase = Out + (size_t)b * DIMC * NPIX;

    #pragma unroll
    for (int t = 0; t < 4; ++t) {
        const int mrow0 = m0 + t * 16 + half * 8;   // C/D layout: VGPR v -> M = v + half*8
        float* orow = obase + (size_t)mrow0 * NPIX + n0 + lr;
        #pragma unroll
        for (int v = 0; v < 8; ++v) {
            float val = accs[t][v] + bias[mrow0 + v];
            if (EPI >= 1) val = gelu_exact(val);
            if (EPI == 2) val += orow[(size_t)v * NPIX];
            orow[(size_t)v * NPIX] = val;
        }
    }
}

// ---------------------------------------------------------------------------
// fp32 -> bf16 weight convert
// ---------------------------------------------------------------------------
__global__ __launch_bounds__(256) void cvt_w_k(const float* __restrict__ in,
                                               bf16_t* __restrict__ out, int n)
{
    int i = blockIdx.x * 256 + threadIdx.x;
    if (i < n) out[i] = (bf16_t)in[i];
}

// ---------------------------------------------------------------------------
// (B,C,HW) fp32 -> (B,HW,C) bf16 transpose+convert (K-major activations)
// ---------------------------------------------------------------------------
__global__ __launch_bounds__(256) void transpose_cvt_k(const float* __restrict__ X,
                                                       bf16_t* __restrict__ XT)
{
    int i = blockIdx.x * 256 + threadIdx.x;
    if (i >= NB * NPIX * DIMC) return;
    int c = i % DIMC;
    int p = (i / DIMC) % NPIX;
    int b = i / (DIMC * NPIX);
    XT[i] = (bf16_t)X[(size_t)b * SLAB + (size_t)c * NPIX + p];
}

// ---------------------------------------------------------------------------
// GroupNorm(1,C) statistics: 8 partial blocks per sample, then finalize.
// ---------------------------------------------------------------------------
__global__ __launch_bounds__(256) void stats_partial_k(const float* __restrict__ X,
                                                       float2* __restrict__ part)
{
    const int b = blockIdx.y;
    const float* p = X + (size_t)b * SLAB;
    float s = 0.f, q = 0.f;
    for (int i = blockIdx.x * 256 + threadIdx.x; i < SLAB; i += 8 * 256) {
        float v = p[i];
        s += v;
        q += v * v;
    }
    __shared__ float sh_s[256];
    __shared__ float sh_q[256];
    sh_s[threadIdx.x] = s;
    sh_q[threadIdx.x] = q;
    __syncthreads();
    for (int off = 128; off > 0; off >>= 1) {
        if ((int)threadIdx.x < off) {
            sh_s[threadIdx.x] += sh_s[threadIdx.x + off];
            sh_q[threadIdx.x] += sh_q[threadIdx.x + off];
        }
        __syncthreads();
    }
    if (threadIdx.x == 0) part[b * 8 + blockIdx.x] = make_float2(sh_s[0], sh_q[0]);
}

__global__ void stats_final_k(const float2* __restrict__ part,
                              float2* __restrict__ st)
{
    int b = threadIdx.x;
    if (b < NB) {
        float s = 0.f, q = 0.f;
        for (int i = 0; i < 8; ++i) {
            s += part[b * 8 + i].x;
            q += part[b * 8 + i].y;
        }
        float mean = s / (float)SLAB;
        float var  = q / (float)SLAB - mean * mean;
        st[b] = make_float2(mean, rsqrtf(var + 1e-5f));
    }
}

// ---------------------------------------------------------------------------
// Fused GN1-apply + GELU + axial shift + transpose to (B,HW,C) bf16.
// Chunk j = c/154 gets shift s = j-2; zero-padded shift (pad after GELU).
// axis_w=1 -> shift along W (x_lr), axis_w=0 -> shift along H (x_td).
// ---------------------------------------------------------------------------
__global__ __launch_bounds__(256) void gn_gelu_shift_k(
    const float* __restrict__ Hb, const float2* __restrict__ st,
    const float* __restrict__ g, const float* __restrict__ be,
    bf16_t* __restrict__ OutT, int axis_w)
{
    int i = blockIdx.x * 256 + threadIdx.x;
    if (i >= NB * NPIX * DIMC) return;
    int c = i % DIMC;
    int p = (i / DIMC) % NPIX;
    int b = i / (DIMC * NPIX);
    int hh = p / WW, wi = p % WW;
    int s = c / 154 - 2;
    int sh = hh, sw = wi;
    if (axis_w) sw = wi - s; else sh = hh - s;
    float val = 0.f;
    if (sw >= 0 && sw < WW && sh >= 0 && sh < HH) {
        float2 ms = st[b];
        float x = Hb[(size_t)b * SLAB + (size_t)c * NPIX + sh * WW + sw];
        x = (x - ms.x) * ms.y * g[c] + be[c];
        val = gelu_exact(x);
    }
    OutT[i] = (bf16_t)val;
}

// ---------------------------------------------------------------------------
// GN2-apply + transpose to (B,HW,C) bf16 (feeds the final GEMM).
// ---------------------------------------------------------------------------
__global__ __launch_bounds__(256) void gn_apply_T_k(
    const float* __restrict__ S, const float2* __restrict__ st,
    const float* __restrict__ g, const float* __restrict__ be,
    bf16_t* __restrict__ OutT)
{
    int i = blockIdx.x * 256 + threadIdx.x;
    if (i >= NB * NPIX * DIMC) return;
    int c = i % DIMC;
    int p = (i / DIMC) % NPIX;
    int b = i / (DIMC * NPIX);
    float2 ms = st[b];
    float x = S[(size_t)b * SLAB + (size_t)c * NPIX + p];
    OutT[i] = (bf16_t)((x - ms.x) * ms.y * g[c] + be[c]);
}

// ---------------------------------------------------------------------------
// Host-side orchestration (all launches on `stream`, scratch in d_ws).
// ---------------------------------------------------------------------------
extern "C" void kernel_launch(void* const* d_in, const int* in_sizes, int n_in,
                              void* d_out, int out_size, void* d_ws, size_t ws_size,
                              hipStream_t stream)
{
    (void)in_sizes; (void)n_in; (void)out_size; (void)ws_size;

    const float* x   = (const float*)d_in[0];
    const float* w1  = (const float*)d_in[1];
    const float* b1  = (const float*)d_in[2];
    const float* g1  = (const float*)d_in[3];
    const float* be1 = (const float*)d_in[4];
    const float* w21 = (const float*)d_in[5];
    const float* b21 = (const float*)d_in[6];
    const float* w22 = (const float*)d_in[7];
    const float* b22 = (const float*)d_in[8];
    const float* g2  = (const float*)d_in[9];
    const float* be2 = (const float*)d_in[10];
    const float* w3  = (const float*)d_in[11];
    const float* b3  = (const float*)d_in[12];

    char* ws = (char*)d_ws;
    size_t off = 0;
    const size_t S_T   = (size_t)NB * NPIX * DIMC * sizeof(bf16_t);  // 38.5 MB
    const size_t S_F32 = (size_t)NB * DIMC * NPIX * sizeof(float);   // 77 MB
    const size_t S_W   = (size_t)DIMC * DIMC * sizeof(bf16_t);       // 1.2 MB

    bf16_t* xT   = (bf16_t*)(ws + off); off += S_T;    // reused later as sT
    float*  hbuf = (float*) (ws + off); off += S_F32;  // reused later as s_pre
    bf16_t* lrT  = (bf16_t*)(ws + off); off += S_T;
    bf16_t* tdT  = (bf16_t*)(ws + off); off += S_T;
    bf16_t* w1b  = (bf16_t*)(ws + off); off += S_W;
    bf16_t* w21b = (bf16_t*)(ws + off); off += S_W;
    bf16_t* w22b = (bf16_t*)(ws + off); off += S_W;
    bf16_t* w3b  = (bf16_t*)(ws + off); off += S_W;
    float2* part = (float2*)(ws + off); off += (size_t)NB * 8 * sizeof(float2);
    float2* st1  = (float2*)(ws + off); off += (size_t)NB * sizeof(float2);
    float2* st2  = (float2*)(ws + off); off += (size_t)NB * sizeof(float2);

    const int NW = DIMC * DIMC;
    const int NE = NB * NPIX * DIMC;
    dim3 blk256(256);
    dim3 gW((NW + 255) / 256);
    dim3 gE((NE + 255) / 256);
    dim3 ggrid(84, NB);    // (12 m_supers x 7 n_groups) per batch, 7 waves/block
    dim3 sgrid(8, NB);

    // weight converts
    cvt_w_k<<<gW, blk256, 0, stream>>>(w1,  w1b,  NW);
    cvt_w_k<<<gW, blk256, 0, stream>>>(w21, w21b, NW);
    cvt_w_k<<<gW, blk256, 0, stream>>>(w22, w22b, NW);
    cvt_w_k<<<gW, blk256, 0, stream>>>(w3,  w3b,  NW);

    // x -> (B,HW,C) bf16
    transpose_cvt_k<<<gE, blk256, 0, stream>>>(x, xT);

    // GEMM1: h = W1 * x + b1  (fp32 spatial)
    gemm_wmma_k<0><<<ggrid, 224, 0, stream>>>(w1b, xT, b1, hbuf);

    // GroupNorm1 stats
    stats_partial_k<<<sgrid, 256, 0, stream>>>(hbuf, part);
    stats_final_k<<<1, 32, 0, stream>>>(part, st1);

    // fused GN1 + GELU + axial shifts -> bf16 K-major operands
    gn_gelu_shift_k<<<gE, blk256, 0, stream>>>(hbuf, st1, g1, be1, lrT, 1);
    gn_gelu_shift_k<<<gE, blk256, 0, stream>>>(hbuf, st1, g1, be1, tdT, 0);

    // GEMM2a: s_pre = gelu(W21*lr + b21); GEMM2b: s_pre += gelu(W22*td + b22)
    gemm_wmma_k<1><<<ggrid, 224, 0, stream>>>(w21b, lrT, b21, hbuf);
    gemm_wmma_k<2><<<ggrid, 224, 0, stream>>>(w22b, tdT, b22, hbuf);

    // GroupNorm2 stats + apply (-> bf16 K-major, reuse xT slot)
    stats_partial_k<<<sgrid, 256, 0, stream>>>(hbuf, part);
    stats_final_k<<<1, 32, 0, stream>>>(part, st2);
    gn_apply_T_k<<<gE, blk256, 0, stream>>>(hbuf, st2, g2, be2, xT);

    // GEMM3: out = W3 * s + b3  (fp32 spatial, straight to d_out)
    gemm_wmma_k<0><<<ggrid, 224, 0, stream>>>(w3b, xT, b3, (float*)d_out);
}
